// ATTModule_44865228374086
// MI455X (gfx1250) — matrix operand order repaired and checked by reading, not compiled, not run on hardware
//
#include <hip/hip_runtime.h>
#include <hip/hip_bf16.h>
#include <cstdint>
#include <cstddef>

#define Bn  1024
#define Cn  1024
#define HWn 196
#define An  8
#define CHn 256

typedef __attribute__((ext_vector_type(16))) __bf16 v16bf;
typedef __attribute__((ext_vector_type(8)))  float  v8f;

union Frag16 { v16bf v; unsigned short s[16]; uint4 q[2]; };

__device__ __forceinline__ unsigned short f2bf(float f) {
  unsigned int u = __float_as_uint(f);
  u += 0x7fffu + ((u >> 16) & 1u);              // round-to-nearest-even
  return (unsigned short)(u >> 16);
}

// ---------------- 1) global average pool: one wave per (b,c) row --------------
__global__ void pool_kernel(const float* __restrict__ feat,
                            float* __restrict__ pooled,
                            unsigned short* __restrict__ pooled_bf) {
  int row  = blockIdx.x * 8 + (threadIdx.x >> 5);   // B*C rows, 8 waves/block
  int lane = threadIdx.x & 31;
  const float4* p = (const float4*)(feat + (size_t)row * HWn); // 49 float4 per row
  float4 v0 = p[lane];
  float s = v0.x + v0.y + v0.z + v0.w;
  if (lane < 17) { float4 v1 = p[lane + 32]; s += v1.x + v1.y + v1.z + v1.w; }
#pragma unroll
  for (int m = 16; m; m >>= 1) s += __shfl_xor(s, m, 32);
  if (lane == 0) {
    float mean = s * (1.0f / (float)HWn);
    pooled[row]    = mean;
    pooled_bf[row] = f2bf(mean);
  }
}

// ---------------- 2) weight conversion fp32 -> bf16, K-contiguous -------------
// fc1t[n][k] with n = a*CH + h, k = c  (B-matrix columns become K-contiguous)
__global__ void cvt_fc1(const float* __restrict__ w, unsigned short* __restrict__ out) {
  unsigned int idx = blockIdx.x * 256u + threadIdx.x;   // A*C*CH = 2M
  unsigned int k = idx & (Cn - 1);
  unsigned int n = idx >> 10;
  unsigned int a = n >> 8;
  unsigned int h = n & (CHn - 1);
  out[idx] = f2bf(w[((size_t)(a * Cn + k)) * CHn + h]); // fc1_w[a][c=k][h]
}
// fc2t[m][k] with m = a*C + c, k = h
__global__ void cvt_fc2(const float* __restrict__ w, unsigned short* __restrict__ out) {
  unsigned int idx = blockIdx.x * 256u + threadIdx.x;   // A*CH*C = 2M
  unsigned int k  = idx & (CHn - 1);
  unsigned int mr = idx >> 8;
  unsigned int a  = mr >> 10;
  unsigned int c  = mr & (Cn - 1);
  out[idx] = f2bf(w[((size_t)(a * CHn + k)) * Cn + c]); // fc2_w[a][h=k][c]
}

// ---------------- 3) gate logits + softmax: one wave per batch row ------------
__global__ void gate_kernel(const float* __restrict__ pooled,
                            const float* __restrict__ fc_w,
                            const float* __restrict__ fc_b,
                            float* __restrict__ gate) {
  int b    = blockIdx.x * 8 + (threadIdx.x >> 5);
  int lane = threadIdx.x & 31;
  float acc[An] = {};
  for (int k = lane; k < Cn; k += 32) {
    float p = pooled[(size_t)b * Cn + k];
#pragma unroll
    for (int a = 0; a < An; ++a) acc[a] += p * fc_w[a * Cn + k];
  }
#pragma unroll
  for (int a = 0; a < An; ++a)
    for (int m = 16; m; m >>= 1) acc[a] += __shfl_xor(acc[a], m, 32);
  if (lane == 0) {
    float mx = -1e30f;
#pragma unroll
    for (int a = 0; a < An; ++a) { acc[a] += fc_b[a]; mx = fmaxf(mx, acc[a]); }
    float sum = 0.f;
#pragma unroll
    for (int a = 0; a < An; ++a) { acc[a] = __expf(acc[a] - mx); sum += acc[a]; }
    float inv = 1.0f / sum;
#pragma unroll
    for (int a = 0; a < An; ++a) gate[(size_t)b * An + a] = acc[a] * inv;
  }
}

// ---------------- 4) fused expert MoE: WMMA bf16 GEMM pair --------------------
// One block per 16-row batch tile; loops experts internally (register-
// accumulated mix -> deterministic, no atomics). Writes 1+sigmoid(mixed).
__global__ __launch_bounds__(256)
void expert_kernel(const unsigned short* __restrict__ pooled_bf,
                   const unsigned short* __restrict__ fc1t,
                   const float* __restrict__ fc1_b,
                   const unsigned short* __restrict__ fc2t,
                   const float* __restrict__ fc2_b,
                   const float* __restrict__ gate,
                   float* __restrict__ scaleBuf) {
  __shared__ unsigned short p_lds[16][Cn + 8];    // pooled tile, bf16, padded
  __shared__ unsigned short h_lds[16][CHn + 8];   // hidden tile, bf16, padded
  __shared__ float g_lds[16];

  const int b0   = blockIdx.x * 16;
  const int tid  = threadIdx.x;
  const int wave = tid >> 5;
  const int lane = tid & 31;
  const int m    = lane & 15;               // A-frag row / B&D column
  const int kb   = (lane >> 4) << 3;        // A-frag K sub-base (0 or 8)
  const int kb2  = (lane >> 4) << 4;        // unused; B-frag also uses kb (0 or 8)
  const int rowoff = (lane >> 4) << 3;      // C/D rows r / r+8

  // stage pooled tile: 16 rows x 1024 bf16 (row-wise b128 copies)
  {
    const uint4* src = (const uint4*)pooled_bf;
    for (int i = tid; i < 2048; i += 256) {
      int r = i >> 7, cq = i & 127;
      ((uint4*)&p_lds[r][0])[cq] = src[(size_t)(b0 + r) * 128 + cq];
    }
  }

  v8f macc[8];                              // mixed accumulators: 8 N-tiles/wave
#pragma unroll
  for (int i = 0; i < 8; ++i) macc[i] = (v8f){};

  for (int a = 0; a < An; ++a) {
    __syncthreads();                        // prior iter done with h_lds/g_lds
    if (tid < 16) g_lds[tid] = gate[(size_t)(b0 + tid) * An + a];

    // ---- GEMM1: h = relu(pooled[16x1024] @ fc1[a][1024x256]) ----
#pragma unroll
    for (int t = 0; t < 2; ++t) {
      int nt  = (wave << 1) + t;            // 16 N-tiles / 8 waves
      int col = (nt << 4) + m;
      const unsigned short* brow = fc1t + ((size_t)(a * CHn + col)) * Cn;
      v8f acc = {};
      for (int kk = 0; kk < Cn; kk += 32) {
        Frag16 fa, fb;
        fa.q[0] = *(const uint4*)&p_lds[m][kk + kb];
        fa.q[1] = *(const uint4*)&p_lds[m][kk + 16 + kb];
        fb.q[0] = *(const uint4*)(brow + kk + kb);
        fb.q[1] = *(const uint4*)(brow + kk + 16 + kb);
        acc = __builtin_amdgcn_wmma_f32_16x16x32_bf16(
            false, fa.v, false, fb.v, (short)0, acc, false, false);
      }
      float bias = fc1_b[a * CHn + col];
#pragma unroll
      for (int r = 0; r < 8; ++r) {
        float vv = acc[r] + bias;
        h_lds[r + rowoff][col] = f2bf(vv > 0.f ? vv : 0.f);
      }
    }
    __syncthreads();

    // ---- GEMM2: expert = h[16x256] @ fc2[a][256x1024], gate-weighted mix ----
    float g[8];
#pragma unroll
    for (int r = 0; r < 8; ++r) g[r] = g_lds[r + rowoff];

#pragma unroll
    for (int i = 0; i < 8; ++i) {
      int nt  = (wave << 3) + i;            // 64 N-tiles / 8 waves
      int col = (nt << 4) + m;
      const unsigned short* brow = fc2t + ((size_t)(a * Cn + col)) * CHn;
      v8f acc = {};
      for (int kk = 0; kk < CHn; kk += 32) {
        Frag16 fa, fb;
        fa.q[0] = *(const uint4*)&h_lds[m][kk + kb];
        fa.q[1] = *(const uint4*)&h_lds[m][kk + 16 + kb];
        fb.q[0] = *(const uint4*)(brow + kk + kb);
        fb.q[1] = *(const uint4*)(brow + kk + 16 + kb);
        acc = __builtin_amdgcn_wmma_f32_16x16x32_bf16(
            false, fa.v, false, fb.v, (short)0, acc, false, false);
      }
      float bias = fc2_b[a * Cn + col];
#pragma unroll
      for (int r = 0; r < 8; ++r)
        macc[i][r] += g[r] * (acc[r] + bias);
    }
  }

  // scale = 1 + sigmoid(mixed)
#pragma unroll
  for (int i = 0; i < 8; ++i) {
    int col = (((wave << 3) + i) << 4) + m;
#pragma unroll
    for (int r = 0; r < 8; ++r) {
      float x = macc[i][r];
      float s = 1.0f + 1.0f / (1.0f + __expf(-x));
      scaleBuf[(size_t)(b0 + r + rowoff) * Cn + col] = s;
    }
  }
}

// ---------------- 5) apply: out = x * scale (float4, memory-bound) ------------
__global__ void apply_kernel(const float* __restrict__ feat,
                             const float* __restrict__ scaleBuf,
                             float* __restrict__ out) {
  unsigned int idx = blockIdx.x * 256u + threadIdx.x;  // float4 index
  unsigned int row = idx / 49u;                        // 196/4 = 49 float4 / row
  float s = scaleBuf[row];
  float4 v = ((const float4*)feat)[idx];
  float4 o = make_float4(v.x * s, v.y * s, v.z * s, v.w * s);
  ((float4*)out)[idx] = o;
}

extern "C" void kernel_launch(void* const* d_in, const int* in_sizes, int n_in,
                              void* d_out, int out_size, void* d_ws, size_t ws_size,
                              hipStream_t stream) {
  const float* feat  = (const float*)d_in[0];
  const float* fc_w  = (const float*)d_in[1];
  const float* fc_b  = (const float*)d_in[2];
  const float* fc1_w = (const float*)d_in[3];
  const float* fc1_b = (const float*)d_in[4];
  const float* fc2_w = (const float*)d_in[5];
  const float* fc2_b = (const float*)d_in[6];
  float* out = (float*)d_out;

  // workspace layout (~18 MB)
  float*          pooled    = (float*)d_ws;                                  // 4 MB
  float*          scaleBuf  = pooled + (size_t)Bn * Cn;                      // 4 MB
  unsigned short* pooled_bf = (unsigned short*)(scaleBuf + (size_t)Bn * Cn); // 2 MB
  unsigned short* fc1t      = pooled_bf + (size_t)Bn * Cn;                   // 4 MB
  unsigned short* fc2t      = fc1t + (size_t)An * Cn * CHn;                  // 4 MB
  float*          gate      = (float*)(fc2t + (size_t)An * CHn * Cn);        // 32 KB

  pool_kernel<<<(Bn * Cn) / 8, 256, 0, stream>>>(feat, pooled, pooled_bf);
  cvt_fc1<<<(An * Cn * CHn) / 256, 256, 0, stream>>>(fc1_w, fc1t);
  cvt_fc2<<<(An * CHn * Cn) / 256, 256, 0, stream>>>(fc2_w, fc2t);
  gate_kernel<<<Bn / 8, 256, 0, stream>>>(pooled, fc_w, fc_b, gate);
  expert_kernel<<<Bn / 16, 256, 0, stream>>>(pooled_bf, fc1t, fc1_b,
                                             fc2t, fc2_b, gate, scaleBuf);
  apply_kernel<<<(Bn * Cn * 49) / 256, 256, 0, stream>>>(feat, scaleBuf, out);
}
